// NSA_15822659518820
// MI455X (gfx1250) — matrix operand order repaired and checked by reading, not compiled, not run on hardware
//
#include <hip/hip_runtime.h>
#include <math.h>

typedef __attribute__((ext_vector_type(16))) _Float16      v16h;
typedef __attribute__((ext_vector_type(8)))  float         v8f;
typedef __attribute__((ext_vector_type(4)))  unsigned int  u32x4;
typedef __attribute__((ext_vector_type(8)))  int           i32x8;
typedef __attribute__((ext_vector_type(4)))  int           i32x4;

#define NB     16      // batch
#define CIN    32
#define COUT   32
#define HIN    384
#define HOUT   192
#define NHID   9
#define NK     4       // number of expert kernels
#define ROWLEN (HOUT * CIN)   // 6144 halves per (y) row of xp, channels innermost

#if defined(__HIP_DEVICE_COMPILE__)
#if defined(__has_builtin)
#if __has_builtin(__builtin_amdgcn_tensor_load_to_lds) && __has_builtin(__builtin_amdgcn_s_wait_tensorcnt)
#define HAVE_TDM 1
#endif
#endif
#endif

// ---------------------------------------------------------------------------
// Kernel 0: zero the pooled-sum accumulator (must be re-zeroed every call).
// ---------------------------------------------------------------------------
__global__ void NSA_zero_kernel(float* __restrict__ p, int n) {
  int i = blockIdx.x * blockDim.x + threadIdx.x;
  if (i < n) p[i] = 0.0f;
}

// ---------------------------------------------------------------------------
// Kernel 1: 2x2 max-pool, fp32 -> f16, layout [b][y][x][ci] (ci innermost),
// plus per-(b,c) sum accumulation for the global average pool.
// grid = NB*HOUT blocks (one output row each), blockDim = HOUT threads (x).
// ---------------------------------------------------------------------------
__global__ void NSA_pool_kernel(const float* __restrict__ xin,
                                _Float16* __restrict__ xp,
                                float* __restrict__ pooled_acc) {
  const int b = blockIdx.x / HOUT;
  const int y = blockIdx.x % HOUT;
  const int x = threadIdx.x;                 // 0..191
  __shared__ float psum[HOUT * 33];          // padded stride to dodge bank conflicts

  __align__(32) _Float16 outv[CIN];
  float localsum[CIN];
  const int y2 = 2 * y, x2 = 2 * x;
#pragma unroll 4
  for (int c = 0; c < CIN; ++c) {
    const float* p0 = xin + (((size_t)(b * CIN + c) * HIN + y2) * HIN + x2);
    float m = fmaxf(fmaxf(p0[0], p0[1]), fmaxf(p0[HIN], p0[HIN + 1]));
    outv[c] = (_Float16)m;
    localsum[c] = m;
  }
  // 64B fully-coalesced store of all 32 channels of this pixel.
  _Float16* dst = xp + ((size_t)(b * HOUT + y) * HOUT + x) * CIN;
  *(v16h*)(dst)      = *(const v16h*)(outv);
  *(v16h*)(dst + 16) = *(const v16h*)(outv + 16);

  for (int c = 0; c < CIN; ++c) psum[x * 33 + c] = localsum[c];
  __syncthreads();
  if (threadIdx.x < CIN) {
    float s = 0.0f;
    for (int t = 0; t < HOUT; ++t) s += psum[t * 33 + threadIdx.x];
    atomicAdd(&pooled_acc[b * CIN + threadIdx.x], s);
  }
}

// ---------------------------------------------------------------------------
// Kernel 2: attention MLP (tiny): pooled mean -> fc1+relu -> fc2 -> softmax/T,
// plus aggregated bias. One block, one thread per sample.
// ---------------------------------------------------------------------------
__global__ void NSA_attention_kernel(const float* __restrict__ pooled_acc,
                                     const float* __restrict__ fc1_w,
                                     const float* __restrict__ fc2_w,
                                     const float* __restrict__ fc2_b,
                                     const float* __restrict__ bias,
                                     const int* __restrict__ temperature,
                                     float* __restrict__ att_out,
                                     float* __restrict__ agg_b) {
  const int b = threadIdx.x;
  if (b >= NB) return;
  const float inv = 1.0f / (float)(HOUT * HOUT);
  float pooled[CIN];
  for (int c = 0; c < CIN; ++c) pooled[c] = pooled_acc[b * CIN + c] * inv;
  float h[NHID];
  for (int i = 0; i < NHID; ++i) {
    float s = 0.0f;
    for (int c = 0; c < CIN; ++c) s += pooled[c] * fc1_w[i * CIN + c];
    h[i] = fmaxf(s, 0.0f);
  }
  const float T = (float)temperature[0];
  float logits[NK], mx = -3.0e38f;
  for (int k = 0; k < NK; ++k) {
    float s = fc2_b[k];
    for (int i = 0; i < NHID; ++i) s += h[i] * fc2_w[k * NHID + i];
    logits[k] = s / T;
    mx = fmaxf(mx, logits[k]);
  }
  float e[NK], denom = 0.0f;
  for (int k = 0; k < NK; ++k) { e[k] = expf(logits[k] - mx); denom += e[k]; }
  for (int k = 0; k < NK; ++k) att_out[b * NK + k] = e[k] / denom;
  for (int o = 0; o < COUT; ++o) {
    float s = 0.0f;
    for (int k = 0; k < NK; ++k) s += (e[k] / denom) * bias[k * COUT + o];
    agg_b[b * COUT + o] = s;
  }
}

// ---------------------------------------------------------------------------
// Kernel 3: aggregate expert weights with attention and shuffle directly into
// a 16-bit A-matrix (16x32) per-lane order, contiguous k (same order as B in the conv):
//   lane L (M = mt*16 + L%16), element j  ->  K(ci) = j + 16*(L>>4)
// afrag layout: [b][tap 0..8][mt 0..1][lane 0..31][j 0..15] halves.
// grid = NB*9*2 blocks of 32 threads.
// ---------------------------------------------------------------------------
__global__ void NSA_afrag_kernel(const float* __restrict__ att,
                                 const float* __restrict__ weight,
                                 _Float16* __restrict__ afrag) {
  const int bid = blockIdx.x;
  const int mt = bid & 1;
  const int t  = (bid >> 1) % 9;
  const int b  = bid / 18;
  const int L  = threadIdx.x;                 // lane 0..31
  const int ky = t / 3, kx = t % 3;
  float a[NK];
  for (int k = 0; k < NK; ++k) a[k] = att[b * NK + k];
  const int m = mt * 16 + (L & 15);
  const int kbase = (L >> 4) * 16;
  __align__(32) _Float16 frag[16];
#pragma unroll
  for (int j = 0; j < 16; ++j) {
    const int ci = j + kbase;
    float s = 0.0f;
    for (int k = 0; k < NK; ++k)
      s += a[k] * weight[(((k * COUT + m) * CIN + ci) * 3 + ky) * 3 + kx];
    frag[j] = (_Float16)s;
  }
  *(v16h*)(afrag + ((((size_t)b * 9 + t) * 2 + mt) * 32 + L) * 16) = *(const v16h*)frag;
}

// ---------------------------------------------------------------------------
// Kernel 4: implicit-GEMM 3x3 conv via WMMA f32_16x16x32_f16 + bias + sigmoid.
// One block = (sample b, output row y); 12 waves x 16 pixels = 192 pixels.
// Three input rows staged into LDS by the Tensor Data Mover.
// ---------------------------------------------------------------------------
__global__ void __launch_bounds__(384)
NSA_conv_kernel(const _Float16* __restrict__ xp,
                const _Float16* __restrict__ afrag,
                const float* __restrict__ agg_b,
                float* __restrict__ out) {
  const int b = blockIdx.x / HOUT;
  const int y = blockIdx.x % HOUT;
  const int wave = threadIdx.x >> 5;          // 0..11
  const int L    = threadIdx.x & 31;
  const int ystart = (y == 0) ? 0 : y - 1;
  const int nvalid = (y == 0 || y == HOUT - 1) ? 2 : 3;
  const int slot0  = (y == 0) ? 1 : 0;        // LDS row slot of first valid row

  __shared__ __align__(32) _Float16 ldsB[3 * ROWLEN];   // 36 KB: rows y-1,y,y+1

  // Zero the out-of-image halo row slot at the vertical boundaries.
  if (y == 0 || y == HOUT - 1) {
    const int zslot = (y == 0) ? 0 : 2;
    v16h zero = {};
    v16h* zp = (v16h*)&ldsB[zslot * ROWLEN];
    if (threadIdx.x < ROWLEN / 16) zp[threadIdx.x] = zero;   // 384 x 32B
  }

  const _Float16* gsrc = xp + ((size_t)(b * HOUT + ystart) * HOUT) * CIN;
  _Float16* ldst = &ldsB[slot0 * ROWLEN];

#ifdef HAVE_TDM
  if (threadIdx.x < 32) {   // single wave issues the DMA
    const unsigned long long ga = (unsigned long long)(uintptr_t)gsrc;
    const unsigned int lds_off  = (unsigned int)(uintptr_t)ldst;
    const unsigned int dim0 = ROWLEN;                 // halves per row (contig)
    const unsigned int dim1 = (unsigned int)(HOUT - ystart);
    u32x4 g0;
    g0.x = 1u;                                        // count=1 valid descriptor
    g0.y = lds_off;                                   // lds_addr (bytes)
    g0.z = (unsigned int)(ga & 0xFFFFFFFFu);          // global_addr[31:0]
    g0.w = (unsigned int)((ga >> 32) & 0x01FFFFFFu) | (2u << 30);  // addr[56:32] | type=2
    i32x8 g1;
    g1[0] = (int)(1u << 16);                          // data_size=1 (2 bytes/elem)
    g1[1] = (int)((dim0 & 0xFFFFu) << 16);            // tensor_dim0[15:0]
    g1[2] = (int)(((dim0 >> 16) & 0xFFFFu) | ((dim1 & 0xFFFFu) << 16)); // dim0 hi | dim1 lo
    g1[3] = (int)(((dim1 >> 16) & 0xFFFFu) | ((dim0 & 0xFFFFu) << 16)); // dim1 hi | tile_dim0
    g1[4] = (int)(unsigned int)nvalid;                // tile_dim1 = rows, tile_dim2 = 0
    g1[5] = (int)dim0;                                // tensor_dim0_stride[31:0]
    g1[6] = 0;
    g1[7] = 0;
    i32x4 gz = {0, 0, 0, 0};
#if defined(__clang_major__) && (__clang_major__ >= 23)
    i32x8 gz8 = {0, 0, 0, 0, 0, 0, 0, 0};
    __builtin_amdgcn_tensor_load_to_lds(g0, g1, gz, gz, gz8, 0);
#else
    __builtin_amdgcn_tensor_load_to_lds(g0, g1, gz, gz, 0);
#endif
    __builtin_amdgcn_s_wait_tensorcnt(0);
  }
#else
  for (int i = threadIdx.x; i < nvalid * (ROWLEN / 16); i += blockDim.x)
    ((v16h*)ldst)[i] = ((const v16h*)gsrc)[i];
#endif
  __syncthreads();

  // Implicit GEMM: acc[32 cout x 16 px] = sum over 9 taps of A(32x32) * B(32x16).
  v8f acc0 = {};
  v8f acc1 = {};
  const int xx0 = wave * 16 + (L & 15);       // this lane's output pixel x (N index)
  const int ci0 = (L >> 4) * 16;              // K half handled by this lane
  const _Float16* af = afrag + (size_t)b * 9 * 2 * 32 * 16;

#pragma unroll
  for (int t = 0; t < 9; ++t) {
    const int ky = t / 3, kx = t % 3;
    const int xx = xx0 + kx - 1;
    v16h bfrag = {};
    if (xx >= 0 && xx < HOUT)
      bfrag = *(const v16h*)&ldsB[(ky * HOUT + xx) * CIN + ci0];   // 32B contiguous
    v16h a0 = *(const v16h*)(af + (((size_t)t * 2 + 0) * 32 + L) * 16);
    v16h a1 = *(const v16h*)(af + (((size_t)t * 2 + 1) * 32 + L) * 16);
    acc0 = __builtin_amdgcn_wmma_f32_16x16x32_f16(false, a0, false, bfrag,
                                                  (short)0, acc0, false, false);
    acc1 = __builtin_amdgcn_wmma_f32_16x16x32_f16(false, a1, false, bfrag,
                                                  (short)0, acc1, false, false);
  }

  // Epilogue: C/D layout -> lanes 0-15: M=v, lanes 16-31: M=8+v; N = L&15.
  const int mrow = (L >> 4) * 8;
  float* ob = out + (size_t)b * COUT * HOUT * HOUT + (size_t)y * HOUT + xx0;
#pragma unroll
  for (int v = 0; v < 8; ++v) {
    const int c0 = mrow + v;            // cout in M-tile 0
    const int c1 = 16 + mrow + v;       // cout in M-tile 1
    float v0 = acc0[v] + agg_b[b * COUT + c0];
    float v1 = acc1[v] + agg_b[b * COUT + c1];
    ob[(size_t)c0 * HOUT * HOUT] = 1.0f / (1.0f + __expf(-v0));
    ob[(size_t)c1 * HOUT * HOUT] = 1.0f / (1.0f + __expf(-v1));
  }
}

// ---------------------------------------------------------------------------
extern "C" void kernel_launch(void* const* d_in, const int* in_sizes, int n_in,
                              void* d_out, int out_size, void* d_ws, size_t ws_size,
                              hipStream_t stream) {
  const float* x      = (const float*)d_in[0];
  const int*   temp   = (const int*)  d_in[1];
  const float* fc1_w  = (const float*)d_in[2];
  const float* fc2_w  = (const float*)d_in[3];
  const float* fc2_b  = (const float*)d_in[4];
  const float* weight = (const float*)d_in[5];
  const float* bias   = (const float*)d_in[6];
  float* out = (float*)d_out;

  char* ws = (char*)d_ws;
  size_t off = 0;
  _Float16* xp = (_Float16*)(ws + off);
  off += (size_t)NB * HOUT * HOUT * CIN * sizeof(_Float16);      // 37,748,736 B
  float* pooled = (float*)(ws + off);
  off += (size_t)NB * CIN * sizeof(float);  off = (off + 255) & ~(size_t)255;
  float* att = (float*)(ws + off);
  off += (size_t)NB * NK * sizeof(float);   off = (off + 255) & ~(size_t)255;
  float* aggb = (float*)(ws + off);
  off += (size_t)NB * COUT * sizeof(float); off = (off + 255) & ~(size_t)255;
  _Float16* afrag = (_Float16*)(ws + off);
  off += (size_t)NB * 9 * 2 * 32 * 16 * sizeof(_Float16);        // 294,912 B
  (void)ws_size; (void)in_sizes; (void)n_in; (void)out_size;

  NSA_zero_kernel<<<2, 256, 0, stream>>>(pooled, NB * CIN);
  NSA_pool_kernel<<<NB * HOUT, HOUT, 0, stream>>>(x, xp, pooled);
  NSA_attention_kernel<<<1, 64, 0, stream>>>(pooled, fc1_w, fc2_w, fc2_b, bias,
                                             temp, att, aggb);
  NSA_afrag_kernel<<<NB * 9 * 2, 32, 0, stream>>>(att, weight, afrag);
  NSA_conv_kernel<<<NB * HOUT, 384, 0, stream>>>(xp, afrag, aggb, out);
}
